// MoELayer_58445914964619
// MI455X (gfx1250) — compile-verified, matrix operand order checked by reading, not run on hardware
//
#include <hip/hip_runtime.h>
#include <math.h>

// MoE dims (fixed by reference)
constexpr int TT = 8192;   // tokens = B*S
constexpr int DD = 1024;   // model dim
constexpr int FF = 4096;   // ffn dim
constexpr int EE = 8;      // experts
constexpr int MT = 16;     // tokens per tile
constexpr int FC = 256;    // F chunk per iteration
constexpr int XS = 1032;   // LDS stride (halves) for X tile, 16B-aligned, bank-padded
constexpr int HS = 264;    // LDS stride (halves) for H tile

typedef __attribute__((ext_vector_type(16))) __bf16 v16bf;
typedef __attribute__((ext_vector_type(8)))  __bf16 v8bf;
typedef __attribute__((ext_vector_type(8)))  float  v8f;
typedef __attribute__((ext_vector_type(4)))  float  v4f;

// Branch-free exact-grade gelu: erf via Abramowitz-Stegun 7.1.26 (|err| < 1.5e-7),
// __expf lowers to a single v_exp_f32 (TRANS, co-executes with XDL WMMA).
__device__ __forceinline__ float gelu_exact(float v) {
    float ax = fabsf(v) * 0.70710678118654752f;
    float t  = 1.0f / fmaf(0.3275911f, ax, 1.0f);
    float p  = t * fmaf(t, fmaf(t, fmaf(t, fmaf(t, 1.061405429f, -1.453152027f),
                                        1.421413741f), -0.284496736f), 0.254829592f);
    float er = 1.0f - p * __expf(-ax * ax);     // erf(|v|/sqrt2)
    return 0.5f * v * (1.0f + copysignf(er, v));
}

// Build a 16-bit A/B WMMA fragment: two contiguous 16B runs per lane
// (works for LDS or global bf16 sources).
__device__ __forceinline__ v16bf ld_frag(const __bf16* p) {
    v8bf lo = *(const v8bf*)p;
    v8bf hi = *(const v8bf*)(p + 16);
    return __builtin_shufflevector(lo, hi, 0,1,2,3,4,5,6,7,8,9,10,11,12,13,14,15);
}

// Convert two 8-float runs (4x v4f) into a bf16 fragment (fallback path only).
__device__ __forceinline__ v16bf cvt_frag(v4f f0, v4f f1, v4f f2, v4f f3) {
    v16bf b;
#pragma unroll
    for (int j = 0; j < 4; ++j) {
        b[j]      = (__bf16)f0[j];
        b[4 + j]  = (__bf16)f1[j];
        b[8 + j]  = (__bf16)f2[j];
        b[12 + j] = (__bf16)f3[j];
    }
    return b;
}

// ---------------- Kernel 0: one-shot fp32 -> bf16 weight conversion ----------------
__global__ __launch_bounds__(256) void cvt_weights(
    const float* __restrict__ w1, const float* __restrict__ w2,
    __bf16* __restrict__ w1b, __bf16* __restrict__ w2b)
{
    const size_t N = (size_t)EE * FF * DD;       // 33,554,432 per tensor
    size_t i = ((size_t)blockIdx.x * blockDim.x + threadIdx.x) * 8;
    if (i >= N) return;
    v4f a0 = *(const v4f*)(w1 + i);
    v4f a1 = *(const v4f*)(w1 + i + 4);
    v4f b0 = *(const v4f*)(w2 + i);
    v4f b1 = *(const v4f*)(w2 + i + 4);
    v8bf oa, ob;
#pragma unroll
    for (int j = 0; j < 4; ++j) {
        oa[j] = (__bf16)a0[j]; oa[4 + j] = (__bf16)a1[j];
        ob[j] = (__bf16)b0[j]; ob[4 + j] = (__bf16)b1[j];
    }
    *(v8bf*)(w1b + i) = oa;
    *(v8bf*)(w2b + i) = ob;
}

// ---------------- Kernel 1: gating (one wave per token) ----------------
__global__ __launch_bounds__(256) void moe_gate(
    const float* __restrict__ x, const float* __restrict__ gw,
    float* __restrict__ probs, int* __restrict__ cnt,
    int* __restrict__ tok_l, int* __restrict__ slot_l, float* __restrict__ wgt_l)
{
    const int lane = threadIdx.x & 31;
    const int wid  = threadIdx.x >> 5;
    const int t    = blockIdx.x * 8 + wid;
    const float* xp = x + (size_t)t * DD;

    float acc[EE];
#pragma unroll
    for (int e = 0; e < EE; ++e) acc[e] = 0.0f;

    for (int i = 0; i < DD / 32; ++i) {
        float xv = xp[i * 32 + lane];
#pragma unroll
        for (int e = 0; e < EE; ++e)
            acc[e] += xv * gw[e * DD + i * 32 + lane];
    }
#pragma unroll
    for (int e = 0; e < EE; ++e) {
#pragma unroll
        for (int m = 16; m >= 1; m >>= 1) acc[e] += __shfl_xor(acc[e], m, 32);
    }
    float mx = acc[0];
#pragma unroll
    for (int e = 1; e < EE; ++e) mx = fmaxf(mx, acc[e]);
    float p[EE]; float sum = 0.0f;
#pragma unroll
    for (int e = 0; e < EE; ++e) { p[e] = expf(acc[e] - mx); sum += p[e]; }
    float inv_s = 1.0f / sum;
#pragma unroll
    for (int e = 0; e < EE; ++e) p[e] *= inv_s;

    int i0 = 0; float v0 = p[0];
#pragma unroll
    for (int e = 1; e < EE; ++e) if (p[e] > v0) { v0 = p[e]; i0 = e; }
    int i1 = (i0 == 0) ? 1 : 0; float v1 = -1.0f;
#pragma unroll
    for (int e = 0; e < EE; ++e) if (e != i0 && p[e] > v1) { v1 = p[e]; i1 = e; }
    float inv = 1.0f / (v0 + v1 + 1e-9f);

    if (lane == 0) {
#pragma unroll
        for (int e = 0; e < EE; ++e) probs[(size_t)t * EE + e] = p[e];
        int p0 = atomicAdd(&cnt[i0], 1);
        tok_l[i0 * TT + p0] = t; slot_l[i0 * TT + p0] = 0; wgt_l[i0 * TT + p0] = v0 * inv;
        int p1 = atomicAdd(&cnt[i1], 1);
        tok_l[i1 * TT + p1] = t; slot_l[i1 * TT + p1] = 1; wgt_l[i1 * TT + p1] = v1 * inv;
    }
}

// ---------------- Kernel 2: gathered expert FFN (WMMA bf16) ----------------
// PRE=true: weights pre-converted to bf16 (pure b128 fragment loads, no cvt VALU).
template <bool PRE>
__global__ __launch_bounds__(256) void moe_expert(
    const float* __restrict__ x,
    const float*  __restrict__ w1f, const __bf16* __restrict__ w1b,
    const float*  __restrict__ b1,
    const float*  __restrict__ w2f, const __bf16* __restrict__ w2b,
    const float*  __restrict__ b2,
    const int* __restrict__ cnt, const int* __restrict__ tok_l,
    const int* __restrict__ slot_l, const float* __restrict__ wgt_l,
    float* __restrict__ slots)
{
    const int e    = blockIdx.y;
    const int nTok = cnt[e];
    const int tile = blockIdx.x;
    if (tile * MT >= nTok) return;

    __shared__ __align__(16) __bf16 Xs[MT * XS];
    __shared__ __align__(16) __bf16 Hs[MT * HS];
    __shared__ int   rowSlot[MT];
    __shared__ int   rowTok[MT];
    __shared__ float rowW[MT];

    const int tid  = threadIdx.x;
    const int lane = tid & 31;
    const int wid  = tid >> 5;       // 0..7
    const int hi   = lane >> 4;      // 0/1 half-wave
    const int nlo  = lane & 15;

    if (tid < MT) {
        int idx = tile * MT + tid;
        if (idx < nTok) {
            int tk = tok_l[e * TT + idx];
            rowTok[tid]  = tk;
            rowSlot[tid] = tk * 2 + slot_l[e * TT + idx];
            rowW[tid]    = wgt_l[e * TT + idx];
        } else {
            rowTok[tid]  = 0;                // safe read source
            rowSlot[tid] = 2 * TT + tid;     // dump rows (never read)
            rowW[tid]    = 0.0f;
        }
    }
    __syncthreads();

    // Gather X tile -> bf16 LDS (row-major, padded). 4096 float4 / 256 thr.
    for (int it = 0; it < 16; ++it) {
        int v  = it * 256 + tid;
        int r  = v >> 8;
        int c4 = v & 255;
        v4f xv = *(const v4f*)(x + (size_t)rowTok[r] * DD + c4 * 4);
        __bf16* dst = &Xs[r * XS + c4 * 4];
#pragma unroll
        for (int j = 0; j < 4; ++j) dst[j] = (__bf16)xv[j];
    }
    __syncthreads();

    v8f y[8];
#pragma unroll
    for (int i = 0; i < 8; ++i)
        y[i] = (v8f){0.f,0.f,0.f,0.f,0.f,0.f,0.f,0.f};

    const float*  w1fe = w1f + (size_t)e * FF * DD;
    const __bf16* w1be = w1b + (size_t)e * FF * DD;
    const float*  w2fe = w2f + (size_t)e * DD * FF;
    const __bf16* w2be = w2b + (size_t)e * DD * FF;

    for (int fc = 0; fc < FF / FC; ++fc) {
        const int fbase = fc * FC;

        // ---- Phase 1: H[16, FC] = gelu(X @ w1^T + b1); wave computes n-tiles 2w, 2w+1
        v8f h0 = (v8f){0.f,0.f,0.f,0.f,0.f,0.f,0.f,0.f};
        v8f h1 = (v8f){0.f,0.f,0.f,0.f,0.f,0.f,0.f,0.f};
        const int r0 = fbase + wid * 32 + nlo;

        for (int kb = 0; kb < DD / 32; ++kb) {
            v16bf a = ld_frag(&Xs[nlo * XS + kb * 32 + hi * 8]);
            int kof = kb * 32 + hi * 8;
            v16bf bA, bB;
            if (PRE) {
                bA = ld_frag(w1be + (size_t)r0 * DD + kof);
                bB = ld_frag(w1be + (size_t)(r0 + 16) * DD + kof);
            } else {
                const float* q0 = w1fe + (size_t)r0 * DD + kof;
                const float* q1 = q0 + (size_t)16 * DD;
                bA = cvt_frag(*(const v4f*)(q0), *(const v4f*)(q0 + 4),
                              *(const v4f*)(q0 + 16), *(const v4f*)(q0 + 20));
                bB = cvt_frag(*(const v4f*)(q1), *(const v4f*)(q1 + 4),
                              *(const v4f*)(q1 + 16), *(const v4f*)(q1 + 20));
            }
            h0 = __builtin_amdgcn_wmma_f32_16x16x32_bf16(false, a, false, bA, (short)0, h0, false, false);
            h1 = __builtin_amdgcn_wmma_f32_16x16x32_bf16(false, a, false, bB, (short)0, h1, false, false);
        }
#pragma unroll
        for (int j = 0; j < 8; ++j) {
            int m  = j + hi * 8;
            float u0 = gelu_exact(h0[j] + b1[e * FF + r0]);
            Hs[m * HS + wid * 32 + nlo] = (__bf16)u0;
            float u1 = gelu_exact(h1[j] + b1[e * FF + r0 + 16]);
            Hs[m * HS + wid * 32 + 16 + nlo] = (__bf16)u1;
        }
        __syncthreads();

        // ---- Phase 2: y[16, 128 per wave] += H @ w2^T (K chunk = FC)
        for (int kk = 0; kk < FC / 32; ++kk) {
            v16bf a = ld_frag(&Hs[nlo * HS + kk * 32 + hi * 8]);
            int fof = fbase + kk * 32 + hi * 8;
#pragma unroll
            for (int nt = 0; nt < 8; ++nt) {
                int drow = wid * 128 + nt * 16 + nlo;
                v16bf b;
                if (PRE) {
                    b = ld_frag(w2be + (size_t)drow * FF + fof);
                } else {
                    const float* q = w2fe + (size_t)drow * FF + fof;
                    b = cvt_frag(*(const v4f*)(q), *(const v4f*)(q + 4),
                                 *(const v4f*)(q + 16), *(const v4f*)(q + 20));
                }
                y[nt] = __builtin_amdgcn_wmma_f32_16x16x32_bf16(false, a, false, b, (short)0, y[nt], false, false);
            }
        }
        __syncthreads();
    }

    // ---- Weighted scatter to per-(token,slot) buffer (deterministic, no atomics)
#pragma unroll
    for (int nt = 0; nt < 8; ++nt) {
#pragma unroll
        for (int j = 0; j < 8; ++j) {
            int m    = j + hi * 8;
            int dcol = wid * 128 + nt * 16 + nlo;
            float val = (y[nt][j] + b2[e * DD + dcol]) * rowW[m];
            slots[(size_t)rowSlot[m] * DD + dcol] = val;
        }
    }
}

// ---------------- Kernel 3: combine the two expert slots ----------------
__global__ __launch_bounds__(256) void moe_combine(
    const float* __restrict__ slots, float* __restrict__ out)
{
    int t  = blockIdx.x;
    int c4 = threadIdx.x;
    v4f a = *(const v4f*)(slots + (size_t)(2 * t) * DD + c4 * 4);
    v4f b = *(const v4f*)(slots + (size_t)(2 * t + 1) * DD + c4 * 4);
    *(v4f*)(out + (size_t)t * DD + c4 * 4) = a + b;
}

extern "C" void kernel_launch(void* const* d_in, const int* in_sizes, int n_in,
                              void* d_out, int out_size, void* d_ws, size_t ws_size,
                              hipStream_t stream) {
    const float* x  = (const float*)d_in[0];
    const float* gw = (const float*)d_in[1];
    const float* w1 = (const float*)d_in[2];
    const float* b1 = (const float*)d_in[3];
    const float* w2 = (const float*)d_in[4];
    const float* b2 = (const float*)d_in[5];

    float* out   = (float*)d_out;                 // [T, D]
    float* probs = out + (size_t)TT * DD;         // [T, E]

    // Workspace layout (bytes):
    //   [0,256)                       counts
    //   [256, +E*T*4)                 tok_l
    //   [+E*T*4)                      slot_l
    //   [+E*T*4)                      wgt_l
    //   [lists_end, +(2T+16)*D*4)     slots
    //   [slots_end, +E*F*D*2)         w1 bf16   (optional, if ws large enough)
    //   [.., +E*D*F*2)                w2 bf16
    char* ws      = (char*)d_ws;
    int*   cnt    = (int*)ws;
    int*   tok_l  = (int*)(ws + 256);
    int*   slot_l = tok_l + EE * TT;
    float* wgt_l  = (float*)(slot_l + EE * TT);
    size_t lists_end = 256 + (size_t)3 * EE * TT * 4;
    float* slots  = (float*)(ws + lists_end);
    size_t slots_end = lists_end + (size_t)(2 * TT + 16) * DD * 4;
    __bf16* w1b = (__bf16*)(ws + slots_end);
    __bf16* w2b = w1b + (size_t)EE * FF * DD;
    size_t need_full = slots_end + (size_t)2 * EE * FF * DD * 2;

    const bool pre = (ws_size >= need_full);

    hipMemsetAsync(cnt, 0, EE * sizeof(int), stream);
    if (pre) {
        const size_t N = (size_t)EE * FF * DD;
        int blocks = (int)((N / 8 + 255) / 256);
        cvt_weights<<<blocks, 256, 0, stream>>>(w1, w2, w1b, w2b);
    }
    moe_gate<<<TT / 8, 256, 0, stream>>>(x, gw, probs, cnt, tok_l, slot_l, wgt_l);
    if (pre) {
        moe_expert<true><<<dim3(TT / MT, EE), 256, 0, stream>>>(
            x, w1, w1b, b1, w2, w2b, b2, cnt, tok_l, slot_l, wgt_l, slots);
    } else {
        moe_expert<false><<<dim3(TT / MT, EE), 256, 0, stream>>>(
            x, w1, w1b, b1, w2, w2b, b2, cnt, tok_l, slot_l, wgt_l, slots);
    }
    moe_combine<<<TT, 256, 0, stream>>>(slots, out);
}
